// GraphTransformerEncoder_75462575391237
// MI455X (gfx1250) — matrix operand order repaired and checked by reading, not compiled, not run on hardware
//
#include <hip/hip_runtime.h>
#include <hip/hip_bf16.h>
#include <stdint.h>

#define N_NODES 20000
#define N_EDGES 200000
#define DIM     256
#define HEADS   4
#define DH      64
#define INNER   256
#define EDIM    512
#define FFD     1024
#define OUTD    128
#define DEPTH   2
#define EPS     1e-5f

typedef __attribute__((ext_vector_type(16))) __bf16 v16bf;
typedef __attribute__((ext_vector_type(8)))  __bf16 v8bf;
typedef __attribute__((ext_vector_type(8)))  float  v8f;

// ---------------- helpers ----------------

__device__ inline __bf16 f2bf(float f) {
    unsigned u = __float_as_uint(f);
    u += 0x7fffu + ((u >> 16) & 1u);            // round-to-nearest-even
    unsigned short h = (unsigned short)(u >> 16);
    return __builtin_bit_cast(__bf16, h);
}

__device__ inline void atomicMaxFloat(float* addr, float val) {
    // IEEE trick: valid because init is -1e30; int ordering for >=0,
    // reversed unsigned ordering for <0.
    if (val >= 0.f) atomicMax((int*)addr, __float_as_int(val));
    else            atomicMin((unsigned int*)addr, __float_as_uint(val));
}

// ---------------- conversion kernels ----------------

__global__ void k_f32_to_bf16(const float* __restrict__ in, __bf16* __restrict__ out, long long n) {
    long long i = (long long)blockIdx.x * blockDim.x + threadIdx.x;
    if (i < n) out[i] = f2bf(in[i]);
}

// W [K,N] f32 row-major  ->  Wt [N,K] bf16 row-major
__global__ void k_convT(const float* __restrict__ W, __bf16* __restrict__ Wt, int K, int N) {
    long long i = (long long)blockIdx.x * blockDim.x + threadIdx.x;
    if (i >= (long long)K * N) return;
    int k = (int)(i / N), n = (int)(i % N);
    Wt[(long long)n * K + k] = f2bf(W[i]);
}

// ---------------- WMMA GEMM: C[M,NC] = A[M,KC] @ Bt[NC,KC]^T + bias ----------------
// bf16 inputs, f32 accumulate via v_wmma_f32_16x16x32_bf16.
// 8 waves per block; each wave computes one 16 x (16*NB) strip of C so one
// A-fragment load feeds NB WMMAs. KC/NC are compile-time so all B-fragment
// addresses collapse to immediates off a single base register (no addr spills).

template <int ACT, int NB, int KC, int NC>  // ACT: 0 = none, 1 = tanh-GELU
__global__ void k_wmma_gemm(const __bf16* __restrict__ A, const __bf16* __restrict__ Bt,
                            const float* __restrict__ bias, float* __restrict__ C,
                            __bf16* __restrict__ Cbf, int M) {
    constexpr int ngroups = NC / (16 * NB);
    int lane = threadIdx.x & 31;
    int wave = threadIdx.x >> 5;
    int ntiles = (M >> 4) * ngroups;
    int tile = blockIdx.x * 8 + wave;
    if (tile >= ntiles) return;                  // whole-wave uniform: EXEC stays all-ones
    int tm = tile / ngroups, g = tile - tm * ngroups;   // tm-major: block's waves share A rows
    int row = lane & 15, half = lane >> 4;

    // A 16-bit 16x32 fragment: lane-half 0 -> K 0..15, half 1 -> K 16..31
    const __bf16* Ar = A  + (size_t)(tm * 16 + row) * KC + half * 16;
    // B 32x16 fragment: lanes 0-15 hold K=0..15 of their column, lanes 16-31 K=16..31
    const __bf16* Br = Bt + (size_t)(g * NB * 16 + row) * KC + half * 16;

    v8f acc[NB];
#pragma unroll
    for (int j = 0; j < NB; j++) acc[j] = (v8f){};

    for (int k0 = 0; k0 < KC; k0 += 32) {
        __builtin_prefetch(Ar + k0 + 64, 0, 0);                      // global_prefetch_b8
        v8bf alo = *(const v8bf*)(Ar + k0);
        v8bf ahi = *(const v8bf*)(Ar + k0 + 8);
        v16bf a;
#pragma unroll
        for (int i = 0; i < 8; i++) { a[i] = alo[i]; a[i + 8] = ahi[i]; }
#pragma unroll
        for (int j = 0; j < NB; j++) {
            v16bf b = *(const v16bf*)(Br + j * 16 * KC + k0);        // imm offsets off one base
            acc[j] = __builtin_amdgcn_wmma_f32_16x16x32_bf16(false, a, false, b, (short)0,
                                                             acc[j], false, false);
        }
    }

    size_t r0 = (size_t)(tm * 16 + half * 8);
#pragma unroll
    for (int j = 0; j < NB; j++) {
        int cj = (g * NB + j) * 16 + row;
        float bv = bias ? bias[cj] : 0.f;
#pragma unroll
        for (int r = 0; r < 8; r++) {
            float v = acc[j][r] + bv;
            if (ACT == 1) {
                float t = v + 0.044715f * v * v * v;
                v = 0.5f * v * (1.f + tanhf(0.7978845608028654f * t));
            }
            if (C)   C[(r0 + r) * NC + cj] = v;
            if (Cbf) Cbf[(r0 + r) * NC + cj] = f2bf(v);
        }
    }
}

// ---------------- LayerNorm (wave32 per row) -> bf16 ----------------

__global__ void k_layernorm_bf16(const float* __restrict__ x, const float* __restrict__ g,
                                 const float* __restrict__ b, __bf16* __restrict__ out, int n) {
    int wave = threadIdx.x >> 5, lane = threadIdx.x & 31;
    int node = blockIdx.x * (blockDim.x >> 5) + wave;
    if (node >= n) return;
    const float* xp = x + (size_t)node * DIM;
    float v[8], s = 0.f, ss = 0.f;
#pragma unroll
    for (int i = 0; i < 8; i++) { v[i] = xp[lane * 8 + i]; s += v[i]; ss += v[i] * v[i]; }
    for (int o = 16; o; o >>= 1) { s += __shfl_xor(s, o, 32); ss += __shfl_xor(ss, o, 32); }
    float mean = s * (1.f / DIM);
    float var  = ss * (1.f / DIM) - mean * mean;
    float rstd = rsqrtf(var + EPS);
    __bf16* op = out + (size_t)node * DIM;
#pragma unroll
    for (int i = 0; i < 8; i++) {
        int c = lane * 8 + i;
        op[c] = f2bf((v[i] - mean) * rstd * g[c] + b[c]);
    }
}

// ---------------- attention ----------------

__global__ void k_attn_init(float* __restrict__ m, float* __restrict__ den,
                            float* __restrict__ agg) {
    int i = blockIdx.x * blockDim.x + threadIdx.x;
    if (i < N_NODES * HEADS) { m[i] = -1e30f; den[i] = 0.f; }
    if (i < N_NODES * INNER) agg[i] = 0.f;
}

__global__ void k_attn_sim(const float* __restrict__ q, const float* __restrict__ kv,
                           const float* __restrict__ ef, const int* __restrict__ src,
                           const int* __restrict__ dst, float* __restrict__ sim,
                           float* __restrict__ m) {
    int idx = blockIdx.x * blockDim.x + threadIdx.x;
    if (idx >= N_EDGES * HEADS) return;
    int e = idx >> 2, h = idx & 3;
    int s = src[e], d = dst[e];
    const float* qp = q  + (size_t)d * INNER + h * DH;
    const float* kp = kv + (size_t)s * (2 * INNER) + h * DH;
    const float* ep = ef + (size_t)e * INNER + h * DH;
    float acc = 0.f;
#pragma unroll 8
    for (int i = 0; i < DH; i++) acc += qp[i] * (kp[i] + ep[i]);
    acc *= 0.125f;                               // DH^-0.5
    sim[idx] = acc;
    atomicMaxFloat(&m[d * HEADS + h], acc);
}

__global__ void k_attn_agg(const float* __restrict__ kv, const float* __restrict__ ef,
                           const int* __restrict__ src, const int* __restrict__ dst,
                           const float* __restrict__ sim, const float* __restrict__ m,
                           float* __restrict__ den, float* __restrict__ agg) {
    int idx = blockIdx.x * blockDim.x + threadIdx.x;
    if (idx >= N_EDGES * HEADS) return;
    int e = idx >> 2, h = idx & 3;
    int s = src[e], d = dst[e];
    float p = __expf(sim[idx] - m[d * HEADS + h]);
    atomicAdd(&den[d * HEADS + h], p);
    const float* vp = kv + (size_t)s * (2 * INNER) + INNER + h * DH;
    const float* ep = ef + (size_t)e * INNER + h * DH;
    float* ap = agg + (size_t)d * INNER + h * DH;
#pragma unroll 8
    for (int i = 0; i < DH; i++) atomicAdd(&ap[i], p * (vp[i] + ep[i]));
}

__global__ void k_attn_norm(const float* __restrict__ agg, const float* __restrict__ den,
                            __bf16* __restrict__ out) {
    int i = blockIdx.x * blockDim.x + threadIdx.x;
    if (i >= N_NODES * INNER) return;
    int n = i >> 8, h = (i >> 6) & 3;
    float dd = den[n * HEADS + h];
    out[i] = f2bf(dd > 0.f ? agg[i] / dd : 0.f);
}

// ---------------- gated residual (wave32 per row), x updated in place ----------------

__global__ void k_gated_residual(const float* __restrict__ y, const float* __restrict__ gw,
                                 float* __restrict__ x, int n) {
    int wave = threadIdx.x >> 5, lane = threadIdx.x & 31;
    int node = blockIdx.x * (blockDim.x >> 5) + wave;
    if (node >= n) return;
    const float* yp = y + (size_t)node * DIM;
    float* xp = x + (size_t)node * DIM;
    float yv[8], rv[8], dot = 0.f;
#pragma unroll
    for (int i = 0; i < 8; i++) {
        int c = lane * 8 + i;
        yv[i] = yp[c]; rv[i] = xp[c];
        dot += yv[i] * gw[c] + rv[i] * gw[DIM + c] + (yv[i] - rv[i]) * gw[2 * DIM + c];
    }
    for (int o = 16; o; o >>= 1) dot += __shfl_xor(dot, o, 32);
    float g = 1.f / (1.f + __expf(-dot));
#pragma unroll
    for (int i = 0; i < 8; i++) {
        int c = lane * 8 + i;
        xp[c] = yv[i] * g + rv[i] * (1.f - g);
    }
}

// ---------------- host ----------------

extern "C" void kernel_launch(void* const* d_in, const int* in_sizes, int n_in,
                              void* d_out, int out_size, void* d_ws, size_t ws_size,
                              hipStream_t stream) {
    const float* x_in      = (const float*)d_in[0];
    const float* edge_attr = (const float*)d_in[1];
    const int*   eidx      = (const int*)d_in[2];
    const float* ln1_g = (const float*)d_in[3];
    const float* ln1_b = (const float*)d_in[4];
    const float* Wq  = (const float*)d_in[5];  const float* bq  = (const float*)d_in[6];
    const float* Wkv = (const float*)d_in[7];  const float* bkv = (const float*)d_in[8];
    const float* We  = (const float*)d_in[9];  const float* be  = (const float*)d_in[10];
    const float* Wo  = (const float*)d_in[11]; const float* bo  = (const float*)d_in[12];
    const float* gateA = (const float*)d_in[13];
    const float* ln2_g = (const float*)d_in[14];
    const float* ln2_b = (const float*)d_in[15];
    const float* Wff1 = (const float*)d_in[16]; const float* bff1 = (const float*)d_in[17];
    const float* Wff2 = (const float*)d_in[18]; const float* bff2 = (const float*)d_in[19];
    const float* gateF = (const float*)d_in[20];
    const float* Wproj = (const float*)d_in[21]; const float* bproj = (const float*)d_in[22];
    const int* src = eidx;
    const int* dst = eidx + N_EDGES;
    float* out = (float*)d_out;

    // -------- workspace carve --------
    uintptr_t p = (uintptr_t)d_ws;
    auto alloc = [&](size_t bytes) -> void* {
        void* r = (void*)p; p += (bytes + 255) & ~(size_t)255; return r;
    };
    __bf16* wq_t   = (__bf16*)alloc((size_t)DEPTH * DIM * INNER * 2);
    __bf16* wkv_t  = (__bf16*)alloc((size_t)DEPTH * DIM * 2 * INNER * 2);
    __bf16* we_t   = (__bf16*)alloc((size_t)DEPTH * EDIM * INNER * 2);
    __bf16* wo_t   = (__bf16*)alloc((size_t)DEPTH * INNER * DIM * 2);
    __bf16* wff1_t = (__bf16*)alloc((size_t)DEPTH * DIM * FFD * 2);
    __bf16* wff2_t = (__bf16*)alloc((size_t)DEPTH * FFD * DIM * 2);
    __bf16* wproj_t= (__bf16*)alloc((size_t)DIM * OUTD * 2);
    __bf16* ea_bf  = (__bf16*)alloc((size_t)N_EDGES * EDIM * 2);
    float*  e_f    = (float*) alloc((size_t)N_EDGES * INNER * 4);
    __bf16* xn_bf  = (__bf16*)alloc((size_t)N_NODES * DIM * 2);
    float*  qbuf   = (float*) alloc((size_t)N_NODES * INNER * 4);
    float*  kvbuf  = (float*) alloc((size_t)N_NODES * 2 * INNER * 4);
    float*  sim    = (float*) alloc((size_t)N_EDGES * HEADS * 4);
    float*  mbuf   = (float*) alloc((size_t)N_NODES * HEADS * 4);
    float*  den    = (float*) alloc((size_t)N_NODES * HEADS * 4);
    float*  agg    = (float*) alloc((size_t)N_NODES * INNER * 4);
    __bf16* agg_bf = (__bf16*)alloc((size_t)N_NODES * INNER * 2);
    float*  ybuf   = (float*) alloc((size_t)N_NODES * DIM * 4);
    __bf16* ff_bf  = (__bf16*)alloc((size_t)N_NODES * FFD * 2);
    float*  x_cur  = (float*) alloc((size_t)N_NODES * DIM * 4);

    auto blks = [](long long n, int t) { return (unsigned)((n + t - 1) / t); };
    auto gblk = [](int M, int ngroups) { return (unsigned)(((M / 16) * ngroups + 7) / 8); };

    // -------- weight conversion (f32 -> bf16, transposed to [N,K]) --------
    for (int d = 0; d < DEPTH; d++) {
        k_convT<<<blks((long long)DIM * INNER, 256), 256, 0, stream>>>(
            Wq + (size_t)d * DIM * INNER, wq_t + (size_t)d * INNER * DIM, DIM, INNER);
        k_convT<<<blks((long long)DIM * 2 * INNER, 256), 256, 0, stream>>>(
            Wkv + (size_t)d * DIM * 2 * INNER, wkv_t + (size_t)d * 2 * INNER * DIM, DIM, 2 * INNER);
        k_convT<<<blks((long long)EDIM * INNER, 256), 256, 0, stream>>>(
            We + (size_t)d * EDIM * INNER, we_t + (size_t)d * INNER * EDIM, EDIM, INNER);
        k_convT<<<blks((long long)INNER * DIM, 256), 256, 0, stream>>>(
            Wo + (size_t)d * INNER * DIM, wo_t + (size_t)d * DIM * INNER, INNER, DIM);
        k_convT<<<blks((long long)DIM * FFD, 256), 256, 0, stream>>>(
            Wff1 + (size_t)d * DIM * FFD, wff1_t + (size_t)d * FFD * DIM, DIM, FFD);
        k_convT<<<blks((long long)FFD * DIM, 256), 256, 0, stream>>>(
            Wff2 + (size_t)d * FFD * DIM, wff2_t + (size_t)d * DIM * FFD, FFD, DIM);
    }
    k_convT<<<blks((long long)DIM * OUTD, 256), 256, 0, stream>>>(Wproj, wproj_t, DIM, OUTD);

    // edge_attr -> bf16
    k_f32_to_bf16<<<blks((long long)N_EDGES * EDIM, 256), 256, 0, stream>>>(
        edge_attr, ea_bf, (long long)N_EDGES * EDIM);

    // running x
    hipMemcpyAsync(x_cur, x_in, (size_t)N_NODES * DIM * 4, hipMemcpyDeviceToDevice, stream);

    const unsigned rowBlocks = blks(N_NODES, 8);          // 8 wave32 rows per 256-thread block
    const unsigned ehBlocks  = blks((long long)N_EDGES * HEADS, 256);
    const unsigned niBlocks  = blks((long long)N_NODES * INNER, 256);

    for (int d = 0; d < DEPTH; d++) {
        // ---- attention block (pre-norm) ----
        k_layernorm_bf16<<<rowBlocks, 256, 0, stream>>>(
            x_cur, ln1_g + d * DIM, ln1_b + d * DIM, xn_bf, N_NODES);
        // q = xn @ Wq + bq            [N,256] x [256,256]
        k_wmma_gemm<0, 8, 256, 256><<<gblk(N_NODES, 2), 256, 0, stream>>>(
            xn_bf, wq_t + (size_t)d * INNER * DIM, bq + d * INNER, qbuf, nullptr, N_NODES);
        // kv = xn @ Wkv + bkv         [N,256] x [256,512]
        k_wmma_gemm<0, 8, 256, 512><<<gblk(N_NODES, 4), 256, 0, stream>>>(
            xn_bf, wkv_t + (size_t)d * 2 * INNER * DIM, bkv + d * 2 * INNER, kvbuf, nullptr,
            N_NODES);
        // e = edge_attr @ We + be     [E,512] x [512,256]  (streaming A)
        k_wmma_gemm<0, 8, 512, 256><<<gblk(N_EDGES, 2), 256, 0, stream>>>(
            ea_bf, we_t + (size_t)d * INNER * EDIM, be + d * INNER, e_f, nullptr, N_EDGES);

        k_attn_init<<<niBlocks, 256, 0, stream>>>(mbuf, den, agg);
        k_attn_sim<<<ehBlocks, 256, 0, stream>>>(qbuf, kvbuf, e_f, src, dst, sim, mbuf);
        k_attn_agg<<<ehBlocks, 256, 0, stream>>>(kvbuf, e_f, src, dst, sim, mbuf, den, agg);
        k_attn_norm<<<niBlocks, 256, 0, stream>>>(agg, den, agg_bf);

        // y = agg @ Wo + bo           [N,256] x [256,256]
        k_wmma_gemm<0, 8, 256, 256><<<gblk(N_NODES, 2), 256, 0, stream>>>(
            agg_bf, wo_t + (size_t)d * DIM * INNER, bo + d * DIM, ybuf, nullptr, N_NODES);
        k_gated_residual<<<rowBlocks, 256, 0, stream>>>(ybuf, gateA + d * 3 * DIM, x_cur, N_NODES);

        // ---- feedforward block (pre-norm) ----
        k_layernorm_bf16<<<rowBlocks, 256, 0, stream>>>(
            x_cur, ln2_g + d * DIM, ln2_b + d * DIM, xn_bf, N_NODES);
        // f = gelu(xn @ Wff1 + bff1)  [N,256] x [256,1024], fused GELU, bf16 out only
        k_wmma_gemm<1, 8, 256, 1024><<<gblk(N_NODES, 8), 256, 0, stream>>>(
            xn_bf, wff1_t + (size_t)d * FFD * DIM, bff1 + d * FFD, nullptr, ff_bf, N_NODES);
        // y = f @ Wff2 + bff2         [N,1024] x [1024,256]
        k_wmma_gemm<0, 8, 1024, 256><<<gblk(N_NODES, 2), 256, 0, stream>>>(
            ff_bf, wff2_t + (size_t)d * DIM * FFD, bff2 + d * DIM, ybuf, nullptr, N_NODES);
        k_gated_residual<<<rowBlocks, 256, 0, stream>>>(ybuf, gateF + d * 3 * DIM, x_cur, N_NODES);
    }

    // ---- final projection ----
    k_f32_to_bf16<<<blks((long long)N_NODES * DIM, 256), 256, 0, stream>>>(
        x_cur, xn_bf, (long long)N_NODES * DIM);
    k_wmma_gemm<0, 8, 256, 128><<<gblk(N_NODES, 1), 256, 0, stream>>>(
        xn_bf, wproj_t, bproj, out, nullptr, N_NODES);
}
